// Decoder_21466246545411
// MI455X (gfx1250) — matrix operand order repaired and checked by reading, not compiled, not run on hardware
//
#include <hip/hip_runtime.h>
#include <hip/hip_bf16.h>
#include <math.h>
#include <stdint.h>

// ---------------- constants (match reference) ----------------
#define VOCAB 1000
#define HID   256
#define VAL   128
#define KEY   128
#define BATCH 128
#define T_ENC 2048
#define T_DEC 256
#define H1    512           // lstm1 hidden
#define K1    (HID + VAL + H1)   // 896  : [emb | ctx | h1]
#define G1    (4 * H1)           // 2048
#define K2    (H1 + KEY)         // 640  : [h1 | h2]
#define G2    (4 * KEY)          // 512
#define K3    (KEY + VAL)        // 256  : [h2 | ctx]
#define G3PAD 1008               // vocab padded to /16

typedef __bf16 bf16_t;
typedef __attribute__((ext_vector_type(16))) __bf16 v16bf;
typedef __attribute__((ext_vector_type(8)))  __bf16 v8bf;
typedef __attribute__((ext_vector_type(8)))  float  v8f;

// ---------------- one-time prep kernels ----------------

// (T_ENC, B, 128) f32  ->  (B, T_ENC, 128) bf16   (L2-resident K/V)
__global__ __launch_bounds__(256) void pack_kv(const float* __restrict__ src,
                                               bf16_t* __restrict__ dst) {
  size_t i = (size_t)blockIdx.x * 256 + threadIdx.x; // over T_ENC*B*128
  int k  = (int)(i & 127);
  size_t r = i >> 7;
  int b  = (int)(r & (BATCH - 1));
  size_t te = r >> 7;
  dst[((size_t)b * T_ENC + te) * 128 + k] = (bf16_t)src[i];
}

// Pack W = [Wa | Wb] (row-major, out = X @ W^T) into per-lane WMMA B-fragment
// layout: dst[((gt*nkc + kc)*32 + lane)*16 + j] = Wcat[g][k]
//   g = gt*16 + (lane&15);  k = kc*32 + (lane>>4)*16 + j
__global__ __launch_bounds__(256) void pack_w(const float* __restrict__ Wa, int Ka,
                                              const float* __restrict__ Wb, int Kb,
                                              int Greal, int nkc,
                                              bf16_t* __restrict__ dst, size_t n) {
  size_t i = (size_t)blockIdx.x * 256 + threadIdx.x;
  if (i >= n) return;
  int j    = (int)(i & 15);
  int lane = (int)((i >> 4) & 31);
  int kc   = (int)((i >> 9) % nkc);
  int gt   = (int)(i / ((size_t)nkc * 512));
  int g = gt * 16 + (lane & 15);
  int k = kc * 32 + (lane >> 4) * 16 + j;
  float v = 0.f;
  if (g < Greal) v = (k < Ka) ? Wa[(size_t)g * Ka + k] : Wb[(size_t)g * Kb + (k - Ka)];
  dst[i] = (bf16_t)v;
}

__global__ __launch_bounds__(256) void bias_sum(const float* a, const float* b,
                                                float* o, int n) {
  int i = blockIdx.x * 256 + threadIdx.x;
  if (i < n) o[i] = a[i] + b[i];
}

__global__ __launch_bounds__(256) void zero_u32(unsigned* p, size_t n) {
  size_t i = (size_t)blockIdx.x * 256 + threadIdx.x;
  if (i < n) p[i] = 0u;
}

// ---------------- per-step kernels ----------------

// xcat1[:, 0:256] = bf16(emb_table[text[b][t]])
__global__ __launch_bounds__(256) void emb_gather(const float* __restrict__ emb,
                                                  const int* __restrict__ text,
                                                  int t, bf16_t* __restrict__ xcat1) {
  int i = blockIdx.x * 256 + threadIdx.x;      // B*HID = 32768
  int b = i >> 8, j = i & 255;
  int tok = text[b * T_DEC + t];
  xcat1[(size_t)b * K1 + j] = (bf16_t)emb[(size_t)tok * HID + j];
}

// out(128 x Greal, stride ldo) = X(128 x K, bf16) @ Wp(packed bf16) [+ bias]
// block = 256 threads = 8 waves; wave w -> row tile m0=w*16; blockIdx.x -> col tile.
// The K x 16 weight panel (shared by all 8 waves) is async-copied to LDS once
// per block (GLOBAL_LOAD_ASYNC_TO_LDS_B128 / ASYNCcnt), then B-fragments are
// served from LDS; A-fragments stream from global (distinct rows per wave).
__global__ __launch_bounds__(256) void wmma_gemm(const bf16_t* __restrict__ X,
                                                 const bf16_t* __restrict__ Wp,
                                                 const float* __restrict__ bias,
                                                 float* __restrict__ out,
                                                 int K, int ldo, int Greal) {
  __shared__ __align__(16) bf16_t wlds[K1 * 16];   // max panel: 896x16 bf16 = 28 KB
  const int tid  = threadIdx.x;
  const int lane = tid & 31;
  const int wave = tid >> 5;
  const int gt   = blockIdx.x;
  const int m0   = wave * 16;
  const int hi   = lane >> 4;
  const int l    = lane & 15;
  const int nkc  = K >> 5;

  // ---- stage weight panel: nkc*1024 bytes, 16B chunks round-robin over 256 thr
  {
    const char* gpanel = (const char*)(Wp + (size_t)gt * nkc * 512);
    const unsigned lbase = (unsigned)(uintptr_t)(void*)wlds; // LDS offset = addr[31:0]
    const int nch = nkc * 64;
    for (int c = tid; c < nch; c += 256) {
      unsigned laddr = lbase + (unsigned)c * 16u;
      unsigned long long gaddr = (unsigned long long)(uintptr_t)(gpanel + (size_t)c * 16);
      asm volatile("global_load_async_to_lds_b128 %0, %1, off"
                   :: "v"(laddr), "v"(gaddr) : "memory");
    }
    asm volatile("s_wait_asynccnt 0x0" ::: "memory");
    __syncthreads();
  }

  const bf16_t* __restrict__ xrow = X + (size_t)(m0 + l) * K + hi * 16;
  const bf16_t* wl = wlds + lane * 16;

  v8f acc = {};
  for (int kc = 0; kc < nkc; ++kc) {
    union { v16bf v; v8bf h[2]; } a, bf;
    a.h[0] = *(const v8bf*)(xrow + kc * 32);
    a.h[1] = *(const v8bf*)(xrow + kc * 32 + 8);
    bf.v   = *(const v16bf*)(wl + (size_t)kc * 512);
    acc = __builtin_amdgcn_wmma_f32_16x16x32_bf16(
        /*neg_a*/false, a.v, /*neg_b*/false, bf.v,
        /*c_mod*/(short)0, acc, /*reuse_a*/false, /*reuse_b*/false);
  }

  int g = gt * 16 + l;
  if (g < Greal) {
    float bv = bias ? bias[g] : 0.f;
#pragma unroll
    for (int r = 0; r < 8; ++r) {
      int row = m0 + hi * 8 + r;               // C layout: VGPR r -> M=r / M=8+r
      out[(size_t)row * ldo + g] = acc[r] + bv;
    }
  }
}

// gates (B x 4H) -> update c, emit bf16 h into up to two concat buffers
__global__ __launch_bounds__(256) void lstm_point(const float* __restrict__ gates,
                                                  const float* __restrict__ bsum,
                                                  float* __restrict__ c, int H,
                                                  bf16_t* __restrict__ d0, int ld0,
                                                  bf16_t* __restrict__ d1, int ld1) {
  int i = blockIdx.x * 256 + threadIdx.x;      // B*H
  int b = i / H, j = i - b * H;
  const float* g4 = gates + (size_t)b * 4 * H;
  float ig = g4[j]         + bsum[j];
  float fg = g4[H + j]     + bsum[H + j];
  float gg = g4[2 * H + j] + bsum[2 * H + j];
  float og = g4[3 * H + j] + bsum[3 * H + j];
  float si = 1.f / (1.f + expf(-ig));
  float sf = 1.f / (1.f + expf(-fg));
  float so = 1.f / (1.f + expf(-og));
  float cn = sf * c[i] + si * tanhf(gg);
  c[i] = cn;
  bf16_t hb = (bf16_t)(so * tanhf(cn));
  d0[(size_t)b * ld0 + j] = hb;
  if (d1) d1[(size_t)b * ld1 + j] = hb;
}

// fused masked-softmax attention; one block (512 thr) per batch element.
// reads h2 (bf16) from xcat3[:,0:128]; writes ctx bf16 into xcat3[:,128:256]
// and xcat1[:,256:384] (for next step's LSTM1 input).
__global__ __launch_bounds__(512) void attn_kernel(const bf16_t* __restrict__ keysT,
                                                   const bf16_t* __restrict__ valuesT,
                                                   const bf16_t* __restrict__ xcat3_h2,
                                                   const int* __restrict__ lens,
                                                   bf16_t* __restrict__ xcat3_ctx,
                                                   bf16_t* __restrict__ xcat1_ctx) {
  __shared__ float h2s[KEY];
  __shared__ float attn[T_ENC];
  __shared__ float sred[512];
  const int b = blockIdx.x, tid = threadIdx.x;

  if (tid < KEY) h2s[tid] = (float)xcat3_h2[(size_t)b * K3 + tid];
  __syncthreads();

  const int len = lens[b];
  float evals[4];
  float lmax = -3.0e38f;
#pragma unroll
  for (int it = 0; it < 4; ++it) {
    int te = tid + it * 512;
    float e = -1.0e30f;
    if (te < len) {
      const v8bf* kp = (const v8bf*)(keysT + ((size_t)b * T_ENC + te) * KEY);
      float acc = 0.f;
#pragma unroll 4
      for (int kk = 0; kk < KEY / 8; ++kk) {
        v8bf kv = kp[kk];
#pragma unroll
        for (int u = 0; u < 8; ++u) acc += (float)kv[u] * h2s[kk * 8 + u];
      }
      e = acc;
    }
    evals[it] = e;
    lmax = fmaxf(lmax, e);
  }
  sred[tid] = lmax; __syncthreads();
  for (int s = 256; s > 0; s >>= 1) {
    if (tid < s) sred[tid] = fmaxf(sred[tid], sred[tid + s]);
    __syncthreads();
  }
  const float gmax = sred[0];
  __syncthreads();

  float lsum = 0.f;
#pragma unroll
  for (int it = 0; it < 4; ++it) {
    float ex = expf(evals[it] - gmax);   // masked entries underflow to 0
    attn[tid + it * 512] = ex;
    lsum += ex;
  }
  sred[tid] = lsum; __syncthreads();
  for (int s = 256; s > 0; s >>= 1) {
    if (tid < s) sred[tid] += sred[tid + s];
    __syncthreads();
  }
  const float inv = 1.f / sred[0];
  __syncthreads();

  // ctx[v] = sum_te attn[te] * V[b][te][v]; 4 te-chunks x 128 v-lanes
  const int v = tid & 127, chunk = tid >> 7;
  const bf16_t* vp = valuesT + ((size_t)b * T_ENC + chunk * 512) * VAL + v;
  float acc = 0.f;
  const float* ap = attn + chunk * 512;
  for (int te = 0; te < 512; ++te) acc += ap[te] * (float)vp[(size_t)te * VAL];
  sred[tid] = acc * inv; __syncthreads();
  if (tid < VAL) {
    float s = sred[tid] + sred[tid + 128] + sred[tid + 256] + sred[tid + 384];
    bf16_t cv = (bf16_t)s;
    xcat3_ctx[(size_t)b * K3 + tid] = cv;
    xcat1_ctx[(size_t)b * K1 + tid] = cv;
  }
}

// ---------------- host driver ----------------
extern "C" void kernel_launch(void* const* d_in, const int* in_sizes, int n_in,
                              void* d_out, int out_size, void* d_ws, size_t ws_size,
                              hipStream_t stream) {
  const float* keys   = (const float*)d_in[0];
  const float* values = (const float*)d_in[1];
  const int*   lens   = (const int*)d_in[2];
  const int*   text   = (const int*)d_in[3];
  const float* emb    = (const float*)d_in[4];
  const float* Wih1   = (const float*)d_in[5];
  const float* Whh1   = (const float*)d_in[6];
  const float* bih1   = (const float*)d_in[7];
  const float* bhh1   = (const float*)d_in[8];
  const float* Wih2   = (const float*)d_in[9];
  const float* Whh2   = (const float*)d_in[10];
  const float* bih2   = (const float*)d_in[11];
  const float* bhh2   = (const float*)d_in[12];
  const float* Wlin   = (const float*)d_in[13];
  const float* blin   = (const float*)d_in[14];
  float* out = (float*)d_out;

  char* ws = (char*)d_ws;
  size_t off = 0;
  auto take = [&](size_t bytes) { char* p = ws + off; off = (off + bytes + 255) & ~(size_t)255; return p; };

  bf16_t* keysT  = (bf16_t*)take((size_t)BATCH * T_ENC * KEY * 2);   // 64 MB (L2-resident)
  bf16_t* valsT  = (bf16_t*)take((size_t)BATCH * T_ENC * VAL * 2);   // 64 MB
  bf16_t* W1p    = (bf16_t*)take((size_t)G1 * K1 * 2);
  bf16_t* W2p    = (bf16_t*)take((size_t)G2 * K2 * 2);
  bf16_t* W3p    = (bf16_t*)take((size_t)G3PAD * K3 * 2);
  float*  bsum1  = (float*)take((size_t)G1 * 4);
  float*  bsum2  = (float*)take((size_t)G2 * 4);
  // state region (zeroed every call, contiguous):
  char*   state0 = ws + off;
  bf16_t* xcat1  = (bf16_t*)take((size_t)BATCH * K1 * 2);  // [emb|ctx|h1]
  bf16_t* xcat2  = (bf16_t*)take((size_t)BATCH * K2 * 2);  // [h1|h2]
  bf16_t* xcat3  = (bf16_t*)take((size_t)BATCH * K3 * 2);  // [h2|ctx]
  float*  c1     = (float*)take((size_t)BATCH * H1 * 4);
  float*  c2     = (float*)take((size_t)BATCH * KEY * 4);
  size_t  stateBytes = (size_t)((ws + off) - state0);
  float*  gates1 = (float*)take((size_t)BATCH * G1 * 4);
  float*  gates2 = (float*)take((size_t)BATCH * G2 * 4);

  // ---- one-time prep ----
  {
    size_t n = (size_t)T_ENC * BATCH * 128;
    pack_kv<<<(unsigned)(n / 256), 256, 0, stream>>>(keys, keysT);
    pack_kv<<<(unsigned)(n / 256), 256, 0, stream>>>(values, valsT);

    size_t n1 = (size_t)G1 * K1;
    pack_w<<<(unsigned)((n1 + 255) / 256), 256, 0, stream>>>(Wih1, HID + VAL, Whh1, H1, G1, K1 / 32, W1p, n1);
    size_t n2 = (size_t)G2 * K2;
    pack_w<<<(unsigned)((n2 + 255) / 256), 256, 0, stream>>>(Wih2, H1, Whh2, KEY, G2, K2 / 32, W2p, n2);
    size_t n3 = (size_t)G3PAD * K3;
    pack_w<<<(unsigned)((n3 + 255) / 256), 256, 0, stream>>>(Wlin, K3, Wlin, 0, VOCAB, K3 / 32, W3p, n3);

    bias_sum<<<G1 / 256, 256, 0, stream>>>(bih1, bhh1, bsum1, G1);
    bias_sum<<<G2 / 256, 256, 0, stream>>>(bih2, bhh2, bsum2, G2);

    size_t nz = stateBytes / 4;
    zero_u32<<<(unsigned)((nz + 255) / 256), 256, 0, stream>>>((unsigned*)state0, nz);
  }

  // ---- sequential decode ----
  for (int t = 0; t < T_DEC; ++t) {
    emb_gather<<<BATCH * HID / 256, 256, 0, stream>>>(emb, text, t, xcat1);

    wmma_gemm<<<G1 / 16, 256, 0, stream>>>(xcat1, W1p, nullptr, gates1, K1, G1, G1);
    lstm_point<<<BATCH * H1 / 256, 256, 0, stream>>>(gates1, bsum1, c1, H1,
                                                     xcat1 + HID + VAL, K1,   // h1 -> next x1
                                                     xcat2, K2);              // h1 -> x2

    wmma_gemm<<<G2 / 16, 256, 0, stream>>>(xcat2, W2p, nullptr, gates2, K2, G2, G2);
    lstm_point<<<BATCH * KEY / 256, 256, 0, stream>>>(gates2, bsum2, c2, KEY,
                                                      xcat2 + H1, K2,         // h2 -> next x2
                                                      xcat3, K3);             // h2 -> pred/attn

    attn_kernel<<<BATCH, 512, 0, stream>>>(keysT, valsT, xcat3, lens,
                                           xcat3 + KEY,                       // ctx -> pred
                                           xcat1 + HID);                      // ctx -> next x1

    wmma_gemm<<<G3PAD / 16, 256, 0, stream>>>(xcat3, W3p, blin,
                                              out + (size_t)t * VOCAB, K3,
                                              T_DEC * VOCAB, VOCAB);
  }
}
